// MinimalMambaLayer_46428596470011
// MI455X (gfx1250) — matrix operand order repaired and checked by reading, not compiled, not run on hardware
//
#include <hip/hip_runtime.h>
#include <hip/hip_bf16.h>

// ---------------------------------------------------------------------------
// Mamba layer for MI455X (gfx1250, wave32).
// GEMMs: v_wmma_f32_16x16x32_f16, double-buffered LDS staged with
// global_load_async_to_lds_b128 (ASYNCcnt). Conv / softplus / scan / gating
// stay fp32 VALU.
// ---------------------------------------------------------------------------

#define D_MODEL   1024
#define D_STATE   16
#define D_CONV    4
#define D_INNER   2048
#define DT_RANK   64
#define SEQ_L     2048
#define BATCH     2
#define TOKENS    (BATCH * SEQ_L)      // 4096

typedef __attribute__((ext_vector_type(16))) _Float16 v16h;
typedef __attribute__((ext_vector_type(8)))  float    v8f;

union F16x16 { float4 f4[2]; v16h v; };

// ---------------------------------------------------------------------------
// async global -> LDS copy (16B per lane), tracked by ASYNCcnt
// ---------------------------------------------------------------------------
__device__ __forceinline__ void async_copy_b128(const void* gsrc, void* ldst) {
    unsigned lds           = (unsigned)(size_t)ldst;       // generic low 32 = LDS addr
    unsigned long long gad = (unsigned long long)(size_t)gsrc;
    asm volatile("global_load_async_to_lds_b128 %0, %1, off"
                 :: "v"(lds), "v"(gad) : "memory");
}
__device__ __forceinline__ void wait_async0() {
    asm volatile("s_wait_asynccnt 0x0" ::: "memory");
}

// ---------------------------------------------------------------------------
// f32 -> f16 conversion
// ---------------------------------------------------------------------------
__global__ void __launch_bounds__(256)
cvt_f32_to_f16(const float* __restrict__ in, _Float16* __restrict__ out, int n) {
    int i = blockIdx.x * 256 + threadIdx.x;
    if (i < n) out[i] = (_Float16)in[i];
}

// dbc[:, :DT_RANK] (stride 96) -> dense f16 (T x 64)
__global__ void __launch_bounds__(256)
cvt_dt_slice(const float* __restrict__ dbc, _Float16* __restrict__ dth) {
    int i = blockIdx.x * 256 + threadIdx.x;   // i < TOKENS*64
    int t = i >> 6, c = i & 63;
    dth[i] = (_Float16)dbc[(size_t)t * 96 + c];
}

// ---------------------------------------------------------------------------
// WMMA GEMM: C[M,N] = A[M,K] * W[N,K]^T   (A,W f16; C f32 accumulate)
// Block tile 128x128, 8 waves (4M x 2N), wave tile 32x64, BK=32.
// Double-buffered LDS, async global->LDS staging.
// EPI: 0 = plain store, 1 = softplus(acc + bias[n]).  NG: N-bound guards.
// ---------------------------------------------------------------------------
#define LDS_STRIDE 40   // 32 f16 + 8 pad: 80B rows (16B aligned, conflict-free)

template<int EPI, bool NG>
__global__ void __launch_bounds__(256)
gemm_wmma(const _Float16* __restrict__ A, const _Float16* __restrict__ W,
          float* __restrict__ Cout, int M, int N, int K,
          const float* __restrict__ bias) {
    __shared__ _Float16 As[2][128 * LDS_STRIDE];
    __shared__ _Float16 Bs[2][128 * LDS_STRIDE];

    const int tid  = threadIdx.x;
    const int wave = tid >> 5;
    const int lane = tid & 31;
    const int wM   = wave & 3;        // 4 waves along M
    const int wN   = wave >> 2;       // 2 waves along N
    const int half = lane >> 4;
    const int r    = lane & 15;

    const int m0 = blockIdx.x * 128;
    const int n0 = blockIdx.y * 128;

    // staging slots: 128 rows x 32 f16 per tile, 16B per lane, 2 iterations
    const int srow = tid >> 2;            // 0..63
    const int sc8  = (tid & 3) * 8;       // f16 column offset

    // zero-fill out-of-range B rows once (both buffers); never rewritten
    if (NG) {
        const float4 z = make_float4(0.f, 0.f, 0.f, 0.f);
        #pragma unroll
        for (int it = 0; it < 2; ++it) {
            int row = srow + it * 64;
            if (n0 + row >= N) {
                *reinterpret_cast<float4*>(&Bs[0][row * LDS_STRIDE + sc8]) = z;
                *reinterpret_cast<float4*>(&Bs[1][row * LDS_STRIDE + sc8]) = z;
            }
        }
    }

    auto stage = [&](int k0, int b) {
        #pragma unroll
        for (int it = 0; it < 2; ++it) {
            int row = srow + it * 64;
            async_copy_b128(A + (size_t)(m0 + row) * K + k0 + sc8,
                            &As[b][row * LDS_STRIDE + sc8]);
        }
        #pragma unroll
        for (int it = 0; it < 2; ++it) {
            int row = srow + it * 64;
            if (!NG || (n0 + row) < N)
                async_copy_b128(W + (size_t)(n0 + row) * K + k0 + sc8,
                                &Bs[b][row * LDS_STRIDE + sc8]);
        }
    };

    v8f acc[2][4] = {};

    stage(0, 0);
    wait_async0();
    __syncthreads();

    int cb = 0;
    for (int k0 = 0; k0 < K; k0 += 32) {
        if (k0 + 32 < K) stage(k0 + 32, cb ^ 1);   // overlap with compute

        // ---- fragments (ISA 7.12.2 16-bit layouts) ----
        v16h afr[2], bfr[4];
        #pragma unroll
        for (int mi = 0; mi < 2; ++mi) {
            const _Float16* base =
                &As[cb][(wM * 32 + mi * 16 + r) * LDS_STRIDE + half * 8];
            F16x16 u;
            u.f4[0] = *reinterpret_cast<const float4*>(base);        // K 0..7 / 8..15
            u.f4[1] = *reinterpret_cast<const float4*>(base + 16);   // K 16..23 / 24..31
            afr[mi] = u.v;
        }
        #pragma unroll
        for (int ni = 0; ni < 4; ++ni) {
            const _Float16* base =
                &Bs[cb][(wN * 64 + ni * 16 + r) * LDS_STRIDE + half * 8];
            F16x16 u;
            u.f4[0] = *reinterpret_cast<const float4*>(base);
            u.f4[1] = *reinterpret_cast<const float4*>(base + 16);
            bfr[ni] = u.v;
        }

        #pragma unroll
        for (int mi = 0; mi < 2; ++mi)
            #pragma unroll
            for (int ni = 0; ni < 4; ++ni)
                acc[mi][ni] = __builtin_amdgcn_wmma_f32_16x16x32_f16(
                    false, afr[mi], false, bfr[ni],
                    (short)0, acc[mi][ni], false, false);

        wait_async0();        // next chunk resident in LDS
        __syncthreads();      // all waves done (their ds reads completed pre-WMMA)
        cb ^= 1;
    }

    // ---- epilogue + store ----
    #pragma unroll
    for (int mi = 0; mi < 2; ++mi) {
        #pragma unroll
        for (int ni = 0; ni < 4; ++ni) {
            int mbase = m0 + wM * 32 + mi * 16 + half * 8;
            int ncol  = n0 + wN * 64 + ni * 16 + r;
            if (!NG || ncol < N) {
                float bv = (EPI == 1) ? bias[ncol] : 0.0f;
                #pragma unroll
                for (int v = 0; v < 8; ++v) {
                    float val = acc[mi][ni][v];
                    if (EPI == 1) {
                        val += bv;                               // + dt_proj_b
                        val = (val > 20.0f) ? val
                             : __logf(1.0f + __expf(val));       // softplus
                    }
                    Cout[(size_t)(mbase + v) * N + ncol] = val;
                }
            }
        }
    }
}

// ---------------------------------------------------------------------------
// Causal depthwise conv (k=4) + bias + SiLU. xz rows are 4096 wide (xi | res).
// ---------------------------------------------------------------------------
__global__ void __launch_bounds__(256)
conv_silu(const float* __restrict__ xz, const float* __restrict__ conv_w,
          const float* __restrict__ conv_b,
          float* __restrict__ xc, _Float16* __restrict__ xch) {
    const int d = blockIdx.x * 256 + threadIdx.x;
    const int t = blockIdx.y;
    const int l = t & (SEQ_L - 1);

    float acc = conv_b[d];
    #pragma unroll
    for (int j = 0; j < D_CONV; ++j) {
        int lj = l - (D_CONV - 1) + j;
        if (lj >= 0)
            acc += conv_w[d * D_CONV + j] *
                   xz[(size_t)(t - (D_CONV - 1) + j) * (2 * D_INNER) + d];
    }
    float s = acc / (1.0f + __expf(-acc));         // SiLU
    xc [(size_t)t * D_INNER + d] = s;
    xch[(size_t)t * D_INNER + d] = (_Float16)s;
}

// ---------------------------------------------------------------------------
// Selective scan fused with D-skip and SiLU(res) gating.
// One channel per thread; 16-state recurrence in registers; B/C via LDS.
// ---------------------------------------------------------------------------
__global__ void __launch_bounds__(256)
selective_scan(const float* __restrict__ delta, const float* __restrict__ xc,
               const float* __restrict__ dbc,   const float* __restrict__ xz,
               const float* __restrict__ A_log, const float* __restrict__ Dp,
               _Float16* __restrict__ y16) {
    const int d = blockIdx.x * 256 + threadIdx.x;  // 0..2047
    const int b = blockIdx.y;

    float a[D_STATE];
    #pragma unroll
    for (int s = 0; s < D_STATE; ++s)
        a[s] = -__expf(A_log[d * D_STATE + s]);    // A = -exp(A_log)
    const float Dd = Dp[d];

    float h[D_STATE];
    #pragma unroll
    for (int s = 0; s < D_STATE; ++s) h[s] = 0.0f;

    __shared__ float sB[D_STATE];
    __shared__ float sC[D_STATE];

    for (int l = 0; l < SEQ_L; ++l) {
        const size_t t = (size_t)b * SEQ_L + l;
        __syncthreads();
        if (threadIdx.x < D_STATE)
            sB[threadIdx.x] = dbc[t * 96 + DT_RANK + threadIdx.x];
        else if (threadIdx.x < 2 * D_STATE)
            sC[threadIdx.x - D_STATE] = dbc[t * 96 + DT_RANK + threadIdx.x];
        __syncthreads();

        const float du   = delta[t * D_INNER + d];
        const float xcv  = xc   [t * D_INNER + d];
        const float resv = xz   [t * (2 * D_INNER) + D_INNER + d];
        const float dx   = du * xcv;

        float y = 0.0f;
        #pragma unroll
        for (int s = 0; s < D_STATE; ++s) {
            float dA = __expf(a[s] * du);
            h[s] = h[s] * dA + dx * sB[s];
            y   += h[s] * sC[s];
        }
        float yv = (y + xcv * Dd) * (resv / (1.0f + __expf(-resv)));
        y16[t * D_INNER + d] = (_Float16)yv;
    }
}

// ---------------------------------------------------------------------------
// Host: pipeline launch (graph-capture safe: kernels only)
// ---------------------------------------------------------------------------
extern "C" void kernel_launch(void* const* d_in, const int* in_sizes, int n_in,
                              void* d_out, int out_size, void* d_ws, size_t ws_size,
                              hipStream_t stream) {
    const float* x         = (const float*)d_in[0];  // (2,2048,1024)
    const float* in_proj_w = (const float*)d_in[1];  // (4096,1024)
    const float* conv_w    = (const float*)d_in[2];  // (2048,1,4)
    const float* conv_b    = (const float*)d_in[3];  // (2048,)
    const float* x_proj_w  = (const float*)d_in[4];  // (96,2048)
    const float* dt_proj_w = (const float*)d_in[5];  // (2048,64)
    const float* dt_proj_b = (const float*)d_in[6];  // (2048,)
    const float* A_log     = (const float*)d_in[7];  // (2048,16)
    const float* Dvec      = (const float*)d_in[8];  // (2048,)
    const float* out_proj_w= (const float*)d_in[9];  // (1024,2048)
    float* out = (float*)d_out;

    char* w = (char*)d_ws;
    size_t o = 0;
    auto alloc = [&](size_t bytes) { void* p = w + o; o = (o + bytes + 255) & ~(size_t)255; return p; };
    _Float16* XH  = (_Float16*)alloc((size_t)TOKENS * D_MODEL * 2);
    _Float16* W1H = (_Float16*)alloc((size_t)2 * D_INNER * D_MODEL * 2);
    _Float16* W2H = (_Float16*)alloc((size_t)96 * D_INNER * 2);
    _Float16* W3H = (_Float16*)alloc((size_t)D_INNER * DT_RANK * 2);
    _Float16* W4H = (_Float16*)alloc((size_t)D_MODEL * D_INNER * 2);
    float*    XZ  = (float*)   alloc((size_t)TOKENS * 2 * D_INNER * 4);
    float*    XC  = (float*)   alloc((size_t)TOKENS * D_INNER * 4);
    _Float16* XCH = (_Float16*)alloc((size_t)TOKENS * D_INNER * 2);
    float*    DBC = (float*)   alloc((size_t)TOKENS * 96 * 4);
    _Float16* DTH = (_Float16*)alloc((size_t)TOKENS * DT_RANK * 2);
    float*    DEL = (float*)   alloc((size_t)TOKENS * D_INNER * 4);
    _Float16* Y16 = (_Float16*)alloc((size_t)TOKENS * D_INNER * 2);

    auto blocks = [](int n) { return (n + 255) / 256; };

    // 1) precision conversion
    cvt_f32_to_f16<<<blocks(TOKENS * D_MODEL),       256, 0, stream>>>(x,          XH,  TOKENS * D_MODEL);
    cvt_f32_to_f16<<<blocks(2 * D_INNER * D_MODEL),  256, 0, stream>>>(in_proj_w,  W1H, 2 * D_INNER * D_MODEL);
    cvt_f32_to_f16<<<blocks(96 * D_INNER),           256, 0, stream>>>(x_proj_w,   W2H, 96 * D_INNER);
    cvt_f32_to_f16<<<blocks(D_INNER * DT_RANK),      256, 0, stream>>>(dt_proj_w,  W3H, D_INNER * DT_RANK);
    cvt_f32_to_f16<<<blocks(D_MODEL * D_INNER),      256, 0, stream>>>(out_proj_w, W4H, D_MODEL * D_INNER);

    // 2) in_proj: xz[4096,4096] = x @ in_proj_w^T
    gemm_wmma<0, false><<<dim3(TOKENS / 128, (2 * D_INNER) / 128), 256, 0, stream>>>(
        XH, W1H, XZ, TOKENS, 2 * D_INNER, D_MODEL, nullptr);

    // 3) causal conv + SiLU
    conv_silu<<<dim3(D_INNER / 256, TOKENS), 256, 0, stream>>>(XZ, conv_w, conv_b, XC, XCH);

    // 4) x_proj: dbc[4096,96] = xc @ x_proj_w^T   (N=96 -> guarded)
    gemm_wmma<0, true><<<dim3(TOKENS / 128, 1), 256, 0, stream>>>(
        XCH, W2H, DBC, TOKENS, 96, D_INNER, nullptr);

    // 5) delta = softplus(dbc[:, :64] @ dt_proj_w^T + b)
    cvt_dt_slice<<<blocks(TOKENS * DT_RANK), 256, 0, stream>>>(DBC, DTH);
    gemm_wmma<1, false><<<dim3(TOKENS / 128, D_INNER / 128), 256, 0, stream>>>(
        DTH, W3H, DEL, TOKENS, D_INNER, DT_RANK, dt_proj_b);

    // 6) selective scan + D-skip + SiLU(res) gating
    selective_scan<<<dim3(D_INNER / 256, BATCH), 256, 0, stream>>>(
        DEL, XC, DBC, XZ, A_log, Dvec, Y16);

    // 7) out_proj: out[4096,1024] = y @ out_proj_w^T
    gemm_wmma<0, false><<<dim3(TOKENS / 128, D_MODEL / 128), 256, 0, stream>>>(
        Y16, W4H, out, TOKENS, D_MODEL, D_INNER, nullptr);

    (void)in_sizes; (void)n_in; (void)out_size; (void)ws_size;
}
